// CLAPQuantizer_13477607375109
// MI455X (gfx1250) — compile-verified, matrix operand order checked by reading, not run on hardware
//
#include <hip/hip_runtime.h>
#include <hip/hip_bf16.h>

// ---------------------------------------------------------------------------
// Residual VQ (CLAP quantizer) for MI455X / gfx1250.
//   inputs : d_in[0] = clap_embeddings f32 [16,2048,512]
//            d_in[1] = codebooks       f32 [8,1024,512]
//   output : d_out   = [ quantized f32 (B*T*D) | indices-as-f32 (B*T*8) | loss ]
// Distance matmul runs on v_wmma_f32_16x16x32_bf16 (f32 accumulate) with two
// independent accumulator chains; residual update / loss use exact f32 rows.
// ---------------------------------------------------------------------------

typedef __attribute__((ext_vector_type(16))) __bf16 v16bf;
typedef __attribute__((ext_vector_type(8)))  __bf16 v8bf;
typedef __attribute__((ext_vector_type(8)))  float  v8f;

#define DIMS   512
#define CODES  1024
#define STAGES 8
#define ROWS   16          // rows handled per wave (= per block)
#define RPAD   516         // LDS row stride in floats (bank spread)

// ---------------------------------------------------------------------------
// Prep A: f32 codebook -> bf16 codebook in workspace
// ---------------------------------------------------------------------------
__global__ void rvq_cvt_cb(const float* __restrict__ cb,
                           __bf16* __restrict__ cbb, int n) {
    int i = blockIdx.x * blockDim.x + threadIdx.x;
    if (i < n) cbb[i] = (__bf16)cb[i];
}

// ---------------------------------------------------------------------------
// Prep B: exact f32 squared norm per code
// ---------------------------------------------------------------------------
__global__ void rvq_cbsq(const float* __restrict__ cb,
                         float* __restrict__ cbsq, int ncodes) {
    int c = blockIdx.x * blockDim.x + threadIdx.x;
    if (c >= ncodes) return;
    const float* row = cb + (size_t)c * DIMS;
    float s = 0.0f;
    for (int k = 0; k < DIMS; k += 4) {
        float4 v = *(const float4*)(row + k);
        s += v.x * v.x + v.y * v.y + v.z * v.z + v.w * v.w;
    }
    cbsq[c] = s;
}

// ---------------------------------------------------------------------------
// Main: one wave (32 threads) per block, 16 rows per wave, all 8 stages.
// ---------------------------------------------------------------------------
__global__ void __launch_bounds__(32)
rvq_main(const float* __restrict__ emb,       // f32 embeddings [N,512]
         const float* __restrict__ cbf,       // f32 codebooks  [8,1024,512]
         const __bf16* __restrict__ cbb,      // bf16 codebooks [8,1024,512]
         const float* __restrict__ cbsq,      // f32 norms      [8,1024]
         float* __restrict__ outQ,            // [N,512]
         float* __restrict__ outI,            // [N,8] indices as float
         float* __restrict__ part)            // [nblocks*8] loss partials
{
    __shared__ __align__(16) float res[ROWS][RPAD];

    const int lane = threadIdx.x;            // 0..31 (wave32)
    const int lm   = lane & 15;              // row / code within half
    const int half = lane >> 4;              // K-half selector
    const int n0   = blockIdx.x * ROWS;      // first row of this block

    // ---- load residual = input rows into LDS (16 f32 per lane per row) ----
    {
        const int k0 = lane * 16;
        #pragma unroll
        for (int m = 0; m < ROWS; ++m) {
            const float* src = emb + (size_t)(n0 + m) * DIMS + k0;
            #pragma unroll
            for (int t = 0; t < 4; ++t) {
                float4 v = *(const float4*)(src + 4 * t);
                *(float4*)&res[m][k0 + 4 * t] = v;
            }
        }
    }
    __syncthreads();

    // ------------------------------ stage loop -----------------------------
    for (int q = 0; q < STAGES; ++q) {
        // Build A fragments: 16 k-blocks of 16x32 bf16.
        // ISA layout: lane (m + 16*h); elements 0..7 <- K = 32*kb + 8h + 0..7,
        //             elements 8..15 <- K = 32*kb + 16 + 8h + 0..7.
        v16bf A[16];
        #pragma unroll
        for (int kb = 0; kb < 16; ++kb) {
            const int b0 = 32 * kb + 8 * half;
            float4 r0 = *(const float4*)&res[lm][b0];
            float4 r1 = *(const float4*)&res[lm][b0 + 4];
            float4 r2 = *(const float4*)&res[lm][b0 + 16];
            float4 r3 = *(const float4*)&res[lm][b0 + 20];
            v16bf a;
            a[0]  = (__bf16)r0.x; a[1]  = (__bf16)r0.y;
            a[2]  = (__bf16)r0.z; a[3]  = (__bf16)r0.w;
            a[4]  = (__bf16)r1.x; a[5]  = (__bf16)r1.y;
            a[6]  = (__bf16)r1.z; a[7]  = (__bf16)r1.w;
            a[8]  = (__bf16)r2.x; a[9]  = (__bf16)r2.y;
            a[10] = (__bf16)r2.z; a[11] = (__bf16)r2.w;
            a[12] = (__bf16)r3.x; a[13] = (__bf16)r3.y;
            a[14] = (__bf16)r3.z; a[15] = (__bf16)r3.w;
            A[kb] = a;
        }

        float best_d[8];
        int   best_i[8];
        #pragma unroll
        for (int j = 0; j < 8; ++j) { best_d[j] = 3.4e38f; best_i[j] = 0; }

        const __bf16* cbq  = cbb  + (size_t)q * CODES * DIMS;
        const float*  csqq = cbsq + (size_t)q * CODES;
        const __bf16* brow0 = cbq + (size_t)lm * DIMS + 8 * half;

        // ------- sweep the 1024 codes in 16-wide chunks -------
        for (int c0 = 0; c0 < CODES / 16; ++c0) {
            const __bf16* brow = brow0 + (size_t)c0 * 16 * DIMS;
            // prefetch next chunk's row into the near (WGP) cache level
            if (c0 + 1 < CODES / 16)
                __builtin_prefetch(brow + (size_t)16 * DIMS, 0, 3);

            // two independent WMMA accumulator chains for XDL-pipe ILP
            v8f acc0 = {};
            v8f acc1 = {};
            #pragma unroll
            for (int kb = 0; kb < 8; ++kb) {
                // chain 0: k-blocks 0..7
                {
                    v8bf lo = *(const v8bf*)(brow + 32 * kb);
                    v8bf hi = *(const v8bf*)(brow + 32 * kb + 16);
                    v16bf b = __builtin_shufflevector(
                        lo, hi, 0, 1, 2, 3, 4, 5, 6, 7,
                        8, 9, 10, 11, 12, 13, 14, 15);
                    acc0 = __builtin_amdgcn_wmma_f32_16x16x32_bf16(
                        false, A[kb], false, b, (short)0, acc0, false, false);
                }
                // chain 1: k-blocks 8..15
                {
                    v8bf lo = *(const v8bf*)(brow + 32 * (kb + 8));
                    v8bf hi = *(const v8bf*)(brow + 32 * (kb + 8) + 16);
                    v16bf b = __builtin_shufflevector(
                        lo, hi, 0, 1, 2, 3, 4, 5, 6, 7,
                        8, 9, 10, 11, 12, 13, 14, 15);
                    acc1 = __builtin_amdgcn_wmma_f32_16x16x32_bf16(
                        false, A[kb + 8], false, b, (short)0, acc1, false, false);
                }
            }
            // dist[m][n] = ||c_n||^2 - 2 <r_m, c_n>; lane holds col n = lm,
            // VGPR j holds row m = j + 8*half.
            const float csq  = csqq[c0 * 16 + lm];
            const int   code = c0 * 16 + lm;
            #pragma unroll
            for (int j = 0; j < 8; ++j) {
                float dot = acc0[j] + acc1[j];
                float dct = fmaf(-2.0f, dot, csq);
                if (dct < best_d[j]) { best_d[j] = dct; best_i[j] = code; }
            }
        }

        // ------- argmin across the 16 lanes of each half (first-match ties) -------
        #pragma unroll
        for (int off = 1; off <= 8; off <<= 1) {
            #pragma unroll
            for (int j = 0; j < 8; ++j) {
                float od = __shfl_xor(best_d[j], off, 32);
                int   oi = __shfl_xor(best_i[j], off, 32);
                if (od < best_d[j] || (od == best_d[j] && oi < best_i[j])) {
                    best_d[j] = od; best_i[j] = oi;
                }
            }
        }

        // ------- gather exact f32 code rows, update residual, loss partial -------
        float ssq = 0.0f;
        const int k0 = lane * 16;
        #pragma unroll
        for (int m = 0; m < ROWS; ++m) {
            const int idx = __shfl(best_i[m & 7], (m >> 3) << 4, 32);
            const float* crow =
                cbf + ((size_t)q * CODES + (size_t)idx) * DIMS + k0;
            #pragma unroll
            for (int t = 0; t < 4; ++t) {
                float4 cv = *(const float4*)(crow + 4 * t);
                float4 rv = *(const float4*)&res[m][k0 + 4 * t];
                rv.x -= cv.x; rv.y -= cv.y; rv.z -= cv.z; rv.w -= cv.w;
                *(float4*)&res[m][k0 + 4 * t] = rv;
                ssq += rv.x * rv.x + rv.y * rv.y + rv.z * rv.z + rv.w * rv.w;
            }
            if (lane == 0)
                outI[(size_t)(n0 + m) * STAGES + q] = (float)idx;
        }
        // wave reduction of the stage sum-of-squares (== stage loss numerator)
        #pragma unroll
        for (int off = 1; off <= 16; off <<= 1)
            ssq += __shfl_xor(ssq, off, 32);
        if (lane == 0)
            part[(size_t)blockIdx.x * STAGES + q] = ssq;
        __syncthreads();  // single-wave WG: lowers to s_nop; keeps LDS ordering explicit
    }

    // ---- quantized = input - final residual ----
    {
        const int k0 = lane * 16;
        #pragma unroll
        for (int m = 0; m < ROWS; ++m) {
            const float* src = emb + (size_t)(n0 + m) * DIMS + k0;
            float* dst = outQ + (size_t)(n0 + m) * DIMS + k0;
            #pragma unroll
            for (int t = 0; t < 4; ++t) {
                float4 iv = *(const float4*)(src + 4 * t);
                float4 rv = *(const float4*)&res[m][k0 + 4 * t];
                float4 ov;
                ov.x = iv.x - rv.x; ov.y = iv.y - rv.y;
                ov.z = iv.z - rv.z; ov.w = iv.w - rv.w;
                *(float4*)(dst + 4 * t) = ov;
            }
        }
    }
}

// ---------------------------------------------------------------------------
// Deterministic fixed-order loss reduction (no float atomics).
// ---------------------------------------------------------------------------
__global__ void rvq_reduce_loss(const float* __restrict__ part,
                                float* __restrict__ outL,
                                int n, float scale) {
    __shared__ float sm[256];
    float s = 0.0f;
    for (int i = threadIdx.x; i < n; i += 256) s += part[i];
    sm[threadIdx.x] = s;
    __syncthreads();
    for (int o = 128; o > 0; o >>= 1) {
        if ((int)threadIdx.x < o) sm[threadIdx.x] += sm[threadIdx.x + o];
        __syncthreads();
    }
    if (threadIdx.x == 0) *outL = sm[0] * scale;
}

// ---------------------------------------------------------------------------
extern "C" void kernel_launch(void* const* d_in, const int* in_sizes, int n_in,
                              void* d_out, int out_size, void* d_ws, size_t ws_size,
                              hipStream_t stream) {
    const float* emb = (const float*)d_in[0];
    const float* cbf = (const float*)d_in[1];

    const int embN = in_sizes[0];          // B*T*D
    const int cbN  = in_sizes[1];          // Q*C*D
    const int N    = embN / DIMS;          // rows

    // workspace layout: [bf16 codebooks | f32 cb_sq | f32 loss partials]
    __bf16* cbb  = (__bf16*)d_ws;
    float*  cbsq = (float*)((char*)d_ws + (size_t)cbN * sizeof(__bf16));
    float*  part = cbsq + STAGES * CODES;

    float* outQ = (float*)d_out;
    float* outI = outQ + embN;
    float* outL = outI + (size_t)N * STAGES;

    const int nblocks = N / ROWS;

    rvq_cvt_cb<<<(cbN + 255) / 256, 256, 0, stream>>>(cbf, cbb, cbN);
    rvq_cbsq<<<(STAGES * CODES + 255) / 256, 256, 0, stream>>>(
        cbf, cbsq, STAGES * CODES);
    rvq_main<<<nblocks, 32, 0, stream>>>(emb, cbf, cbb, cbsq, outQ, outI, part);
    rvq_reduce_loss<<<1, 256, 0, stream>>>(
        part, outL, nblocks * STAGES,
        1.0f / ((float)N * (float)DIMS));
}
